// Model_72069551227167
// MI455X (gfx1250) — compile-verified, matrix operand order checked by reading, not run on hardware
//
#include <hip/hip_runtime.h>
#include <math.h>

// ---------------- model constants ----------------
#define B_  8
#define L_  192
#define Pp_ 96
#define C_  12
#define D_  128
#define HM_ 32
#define H_  4
#define NL_ 2
#define T_  288          // L_+Pp_
#define K_  3456         // T_*C_
#define INV_SCALE 0.08838834764831845f   // 1/sqrt(128)

typedef __attribute__((ext_vector_type(16))) _Float16 v16h;
typedef __attribute__((ext_vector_type(8)))  _Float16 v8h;
typedef __attribute__((ext_vector_type(8)))  float    v8f;

// =====================================================================
// Weight transpose + f32->f16:  Wt[n*K + k] = (f16) W[k*N + n]
// =====================================================================
__global__ void k_w16t(const float* __restrict__ W, _Float16* __restrict__ Wt,
                       int Kd, int N) {
  int i = blockIdx.x * blockDim.x + threadIdx.x;
  if (i >= Kd * N) return;
  int n = i / Kd, k = i % Kd;
  Wt[i] = (_Float16)W[k * N + n];
}

__global__ void k_f32_to_f16(const float* __restrict__ in,
                             _Float16* __restrict__ out, int n) {
  int i = blockIdx.x * blockDim.x + threadIdx.x;
  if (i < n) out[i] = (_Float16)in[i];
}

// =====================================================================
// Periodic MLP: per (t,c) 2->32->1 MLP on (sin,cos) phase. Batch-invariant.
// =====================================================================
__global__ void k_periodic(const float* __restrict__ p_w1, const float* __restrict__ p_b1,
                           const float* __restrict__ p_w2, const float* __restrict__ p_b2,
                           float* __restrict__ per) {
  int i = blockIdx.x * blockDim.x + threadIdx.x;
  if (i >= T_ * C_) return;
  int t = i / C_, c = i % C_;
  float tv = (t < L_) ? (float)t / (float)L_ : (float)(t - L_) / (float)Pp_;
  float ph = 6.283185307179586f * tv;
  float si = sinf(ph), co = cosf(ph);
  float acc = p_b2[c];
  for (int j = 0; j < HM_; ++j) {
    float h = si * p_w1[(c * 2 + 0) * HM_ + j] + co * p_w1[(c * 2 + 1) * HM_ + j]
            + p_b1[c * HM_ + j];
    h = fmaxf(h, 0.f);
    acc += h * p_w2[c * HM_ + j];
  }
  per[i] = acc;
}

// =====================================================================
// Initial node / edge embeddings
// =====================================================================
__global__ void k_temb(const float* __restrict__ time_w, const float* __restrict__ time_b,
                       float* __restrict__ Temb) {
  int i = blockIdx.x * blockDim.x + threadIdx.x;
  if (i >= B_ * T_ * D_) return;
  int d = i % D_, t = (i / D_) % T_;
  float tv = (t < L_) ? (float)t / (float)L_ : (float)(t - L_) / (float)Pp_;
  Temb[i] = tv * time_w[d] + time_b[d];
}

__global__ void k_cemb(const float* __restrict__ chan_w, const float* __restrict__ chan_b,
                       float* __restrict__ Cemb) {
  int i = blockIdx.x * blockDim.x + threadIdx.x;
  if (i >= B_ * C_ * D_) return;
  int d = i % D_, c = (i / D_) % C_;
  Cemb[i] = chan_w[c * D_ + d] + chan_b[d];
}

__global__ void k_u0(const float* __restrict__ x, const float* __restrict__ per,
                     const float* __restrict__ edge_w, const float* __restrict__ edge_b,
                     float* __restrict__ U) {
  long i = (long)blockIdx.x * blockDim.x + threadIdx.x;
  if (i >= (long)B_ * K_ * D_) return;
  int d = (int)(i % D_);
  long bk = i / D_;
  int k = (int)(bk % K_);
  int b = (int)(bk / K_);
  int t = k / C_, c = k % C_;
  float xr = 0.f, ym = 1.f;
  if (t < L_) { xr = x[((long)b * L_ + t) * C_ + c] - per[t * C_ + c]; ym = 0.f; }
  U[i] = xr * edge_w[d] + ym * edge_w[D_ + d] + edge_b[d];
}

// =====================================================================
// Build concatenated f16 activation matrices
//   mode 0: [U | T_emb[k/C]] (2D)     mode 1: [U | C_emb[k%C]] (2D)
// =====================================================================
__global__ void k_concat2(const float* __restrict__ U, const float* __restrict__ other,
                          _Float16* __restrict__ out, int mode) {
  long i = (long)blockIdx.x * blockDim.x + threadIdx.x;
  if (i >= (long)B_ * K_ * 256) return;
  int d2 = (int)(i % 256);
  long bk = i / 256;
  int k = (int)(bk % K_);
  int b = (int)(bk / K_);
  float v;
  if (d2 < D_) v = U[bk * D_ + d2];
  else {
    int idx = mode ? (k % C_) : (k / C_);
    int stride = mode ? C_ : T_;
    v = other[((long)b * stride + idx) * D_ + (d2 - D_)];
  }
  out[i] = (_Float16)v;
}

__global__ void k_concat3(const float* __restrict__ U, const float* __restrict__ qt,
                          const float* __restrict__ qc, _Float16* __restrict__ out) {
  long i = (long)blockIdx.x * blockDim.x + threadIdx.x;
  if (i >= (long)B_ * K_ * 384) return;
  int d3 = (int)(i % 384);
  long bk = i / 384;
  int k = (int)(bk % K_);
  int b = (int)(bk / K_);
  float v;
  if (d3 < D_)        v = U[bk * D_ + d3];
  else if (d3 < 256)  v = qt[((long)b * T_ + k / C_) * D_ + (d3 - D_)];
  else                v = qc[((long)b * C_ + k % C_) * D_ + (d3 - 256)];
  out[i] = (_Float16)v;
}

// =====================================================================
// WMMA GEMM: out[M,N] = act(A16[M,Kd] @ Wt16[N,Kd]^T + bias) (+ res)
// One wave computes a 16x64 strip (4 N-tiles): A fragment loaded once per
// k-step, amortized over 4 v_wmma_f32_16x16x32_f16. A/B lane layout per
// CDNA5 ISA 7.12.2 (16-bit A 16x32): lane<16 holds K chunks
// [k0..k0+7],[k0+16..k0+23]; lane>=16 holds [+8],[+24].
// =====================================================================
__global__ __launch_bounds__(32) void k_gemm16(
    const _Float16* __restrict__ A, const _Float16* __restrict__ Wt,
    const float* __restrict__ bias, const float* __restrict__ res,
    float* __restrict__ out, int M, int Kd, int N, int relu) {
  int lane = threadIdx.x;
  int mt = blockIdx.x * 16;
  int nt0 = blockIdx.y * 64;           // 4 tiles of 16 columns
  int hi = lane >> 4;                  // 0 or 1
  int l15 = lane & 15;
  const _Float16* arow = A  + (long)(mt + l15) * Kd + hi * 8;
  const _Float16* brow = Wt + (long)(nt0 + l15) * Kd + hi * 8;
  const long bstride = 16l * Kd;       // one N-tile of weight rows
  v8f acc[4] = {};
  for (int k0 = 0; k0 < Kd; k0 += 32) {
    v8h a0 = *(const v8h*)(arow + k0);
    v8h a1 = *(const v8h*)(arow + k0 + 16);
    v16h a;
#pragma unroll
    for (int i = 0; i < 8; ++i) { a[i] = a0[i]; a[8 + i] = a1[i]; }
#pragma unroll
    for (int n = 0; n < 4; ++n) {
      v8h b0 = *(const v8h*)(brow + n * bstride + k0);
      v8h b1 = *(const v8h*)(brow + n * bstride + k0 + 16);
      v16h b;
#pragma unroll
      for (int i = 0; i < 8; ++i) { b[i] = b0[i]; b[8 + i] = b1[i]; }
      acc[n] = __builtin_amdgcn_wmma_f32_16x16x32_f16(false, a, false, b,
                                                      (short)0, acc[n], false, false);
    }
  }
#pragma unroll
  for (int n = 0; n < 4; ++n) {
    int ncol = nt0 + n * 16 + l15;
    float bv = bias ? bias[ncol] : 0.f;
#pragma unroll
    for (int r = 0; r < 8; ++r) {
      int m = mt + r + hi * 8;
      if (m < M && ncol < N) {
        float v = acc[n][r] + bv;
        if (relu) v = fmaxf(v, 0.f);
        if (res)  v += res[(long)m * N + ncol];
        out[(long)m * N + ncol] = v;
      }
    }
  }
}

// =====================================================================
// Block-sparse attention. Side 0: query channel c over its T_ edges.
// One block per (b,c); wave h = head h; lane-split over t. +q residual.
// =====================================================================
__global__ __launch_bounds__(128) void k_attn_c(
    const float* __restrict__ q, const float* __restrict__ ks,
    const float* __restrict__ vs, float* __restrict__ outO) {
  int bc = blockIdx.x;
  int b = bc / C_, c = bc % C_;
  int h = threadIdx.x >> 5, lane = threadIdx.x & 31;
  __shared__ float P[H_][T_];
  const float* qv = q + (long)bc * D_ + h * 32;
  float sc[T_ / 32];
  float smax = -1e30f;
#pragma unroll
  for (int i = 0; i < T_ / 32; ++i) {
    int t = lane + i * 32;
    const float* kv = ks + ((long)b * K_ + (long)t * C_ + c) * D_ + h * 32;
    float s = 0.f;
    for (int j = 0; j < 32; ++j) s += qv[j] * kv[j];
    s *= INV_SCALE;
    sc[i] = s; smax = fmaxf(smax, s);
  }
  for (int o = 16; o > 0; o >>= 1) smax = fmaxf(smax, __shfl_xor(smax, o, 32));
  float ssum = 0.f;
#pragma unroll
  for (int i = 0; i < T_ / 32; ++i) { sc[i] = __expf(sc[i] - smax); ssum += sc[i]; }
  for (int o = 16; o > 0; o >>= 1) ssum += __shfl_xor(ssum, o, 32);
  float inv = 1.f / ssum;
#pragma unroll
  for (int i = 0; i < T_ / 32; ++i) P[h][lane + i * 32] = sc[i] * inv;
  __syncthreads();
  float acc = 0.f;
  for (int t = 0; t < T_; ++t)
    acc += P[h][t] * vs[((long)b * K_ + (long)t * C_ + c) * D_ + h * 32 + lane];
  outO[(long)bc * D_ + h * 32 + lane] = qv[lane] + acc;
}

// Side 1: query time t over its C_ edges. One block per (b,t).
__global__ __launch_bounds__(128) void k_attn_t(
    const float* __restrict__ q, const float* __restrict__ ks,
    const float* __restrict__ vs, float* __restrict__ outO) {
  int bt = blockIdx.x;
  int b = bt / T_, t = bt % T_;
  int h = threadIdx.x >> 5, lane = threadIdx.x & 31;
  __shared__ float P[H_][C_];
  const float* qv = q + (long)bt * D_ + h * 32;
  float s = -1e30f;
  if (lane < C_) {
    const float* kv = ks + ((long)b * K_ + (long)t * C_ + lane) * D_ + h * 32;
    float a = 0.f;
    for (int j = 0; j < 32; ++j) a += qv[j] * kv[j];
    s = a * INV_SCALE;
  }
  float smax = s;
  for (int o = 16; o > 0; o >>= 1) smax = fmaxf(smax, __shfl_xor(smax, o, 32));
  float e = (lane < C_) ? __expf(s - smax) : 0.f;
  float ssum = e;
  for (int o = 16; o > 0; o >>= 1) ssum += __shfl_xor(ssum, o, 32);
  if (lane < C_) P[h][lane] = e / ssum;
  __syncthreads();
  float acc = 0.f;
  for (int cc = 0; cc < C_; ++cc)
    acc += P[h][cc] * vs[((long)b * K_ + (long)t * C_ + cc) * D_ + h * 32 + lane];
  outO[(long)bt * D_ + h * 32 + lane] = qv[lane] + acc;
}

// =====================================================================
// LayerNorm over D=128 (one block per row; may run in place)
// =====================================================================
__global__ __launch_bounds__(128) void k_ln(const float* __restrict__ x,
                                            const float* __restrict__ g,
                                            const float* __restrict__ bb,
                                            float* __restrict__ y) {
  int r = blockIdx.x, d = threadIdx.x;
  __shared__ float red[128];
  float v = x[(long)r * D_ + d];
  red[d] = v; __syncthreads();
  for (int o = 64; o > 0; o >>= 1) { if (d < o) red[d] += red[d + o]; __syncthreads(); }
  float mean = red[0] / (float)D_; __syncthreads();
  float dv = v - mean;
  red[d] = dv * dv; __syncthreads();
  for (int o = 64; o > 0; o >>= 1) { if (d < o) red[d] += red[d + o]; __syncthreads(); }
  float var = red[0] / (float)D_;
  y[(long)r * D_ + d] = dv * rsqrtf(var + 1e-5f) * g[d] + bb[d];
}

// =====================================================================
// Final: o_in = [U | T_emb[it] | C_emb[ic]] @ out_w + out_b + periodic
// =====================================================================
__global__ void k_out(const float* __restrict__ U, const float* __restrict__ Temb,
                      const float* __restrict__ Cemb, const float* __restrict__ out_w,
                      const float* __restrict__ out_b, const float* __restrict__ per,
                      float* __restrict__ y) {
  int i = blockIdx.x * blockDim.x + threadIdx.x;
  if (i >= B_ * K_) return;
  int b = i / K_, k = i % K_;
  int t = k / C_, c = k % C_;
  const float* u = U + (long)i * D_;
  const float* te = Temb + ((long)b * T_ + t) * D_;
  const float* ce = Cemb + ((long)b * C_ + c) * D_;
  float acc = out_b[0];
  for (int d = 0; d < D_; ++d)
    acc += u[d] * out_w[d] + te[d] * out_w[D_ + d] + ce[d] * out_w[2 * D_ + d];
  y[((long)b * T_ + t) * C_ + c] = acc + per[t * C_ + c];
}

// =====================================================================
// Host orchestration
// =====================================================================
extern "C" void kernel_launch(void* const* d_in, const int* in_sizes, int n_in,
                              void* d_out, int out_size, void* d_ws, size_t ws_size,
                              hipStream_t stream) {
  (void)in_sizes; (void)n_in; (void)out_size; (void)ws_size;
  const float* x        = (const float*)d_in[0];
  const float* p_w1     = (const float*)d_in[1];
  const float* p_b1     = (const float*)d_in[2];
  const float* p_w2     = (const float*)d_in[3];
  const float* p_b2     = (const float*)d_in[4];
  const float* edge_w   = (const float*)d_in[5];
  const float* edge_b   = (const float*)d_in[6];
  const float* chan_w   = (const float*)d_in[7];
  const float* chan_b   = (const float*)d_in[8];
  const float* time_w   = (const float*)d_in[9];
  const float* time_b   = (const float*)d_in[10];
  const float* att_q_w  = (const float*)d_in[11];
  const float* att_q_b  = (const float*)d_in[12];
  const float* att_k_w  = (const float*)d_in[13];
  const float* att_k_b  = (const float*)d_in[14];
  const float* att_v_w  = (const float*)d_in[15];
  const float* att_v_b  = (const float*)d_in[16];
  const float* att_o_w  = (const float*)d_in[17];
  const float* att_o_b  = (const float*)d_in[18];
  const float* ln0_g    = (const float*)d_in[19];
  const float* ln0_b    = (const float*)d_in[20];
  const float* ln1_g    = (const float*)d_in[21];
  const float* ln1_b    = (const float*)d_in[22];
  const float* edge_nn_w= (const float*)d_in[23];
  const float* edge_nn_b= (const float*)d_in[24];
  const float* out_w    = (const float*)d_in[25];
  const float* out_b    = (const float*)d_in[26];

  char* ws = (char*)d_ws;
  size_t off = 0;
  auto alloc = [&](size_t bytes) -> void* {
    off = (off + 255) & ~(size_t)255;
    void* p = ws + off; off += bytes; return p;
  };
  float* per   = (float*)alloc(sizeof(float) * T_ * C_);
  float* U     = (float*)alloc(sizeof(float) * (size_t)B_ * K_ * D_);
  float* Temb  = (float*)alloc(sizeof(float) * (size_t)B_ * T_ * D_);
  float* Cemb  = (float*)alloc(sizeof(float) * (size_t)B_ * C_ * D_);
  float* qcb   = (float*)alloc(sizeof(float) * (size_t)B_ * C_ * D_);
  float* qtb   = (float*)alloc(sizeof(float) * (size_t)B_ * T_ * D_);
  float* ksb   = (float*)alloc(sizeof(float) * (size_t)B_ * K_ * D_);
  float* vsb   = (float*)alloc(sizeof(float) * (size_t)B_ * K_ * D_);
  float* qproj = (float*)alloc(sizeof(float) * (size_t)B_ * T_ * D_);
  float* attO  = (float*)alloc(sizeof(float) * (size_t)B_ * T_ * D_);
  _Float16* big16   = (_Float16*)alloc(sizeof(_Float16) * (size_t)B_ * K_ * 384);
  _Float16* small16 = (_Float16*)alloc(sizeof(_Float16) * (size_t)B_ * T_ * D_);
  _Float16* w16     = (_Float16*)alloc(sizeof(_Float16) * 491520);

  auto qwt = [&](int l, int s) { return w16 + (size_t)(l * 2 + s) * 98304; };
  auto kwt = [&](int l, int s) { return qwt(l, s) + 16384; };
  auto vwt = [&](int l, int s) { return kwt(l, s) + 32768; };
  auto owt = [&](int l, int s) { return vwt(l, s) + 32768; };
  _Float16* edget = w16 + 4 * 98304;

  // ---- weights -> transposed f16 ----
  for (int l = 0; l < NL_; ++l) for (int s = 0; s < 2; ++s) {
    int ls = l * 2 + s;
    k_w16t<<<(128 * 128 + 255) / 256, 256, 0, stream>>>(att_q_w + (size_t)ls * 128 * 128, qwt(l, s), 128, 128);
    k_w16t<<<(256 * 128 + 255) / 256, 256, 0, stream>>>(att_k_w + (size_t)ls * 256 * 128, kwt(l, s), 256, 128);
    k_w16t<<<(256 * 128 + 255) / 256, 256, 0, stream>>>(att_v_w + (size_t)ls * 256 * 128, vwt(l, s), 256, 128);
    k_w16t<<<(128 * 128 + 255) / 256, 256, 0, stream>>>(att_o_w + (size_t)ls * 128 * 128, owt(l, s), 128, 128);
  }
  for (int l = 0; l < NL_; ++l)
    k_w16t<<<(384 * 128 + 255) / 256, 256, 0, stream>>>(edge_nn_w + (size_t)l * 384 * 128, edget + (size_t)l * 49152, 384, 128);

  // ---- init ----
  k_periodic<<<(T_ * C_ + 127) / 128, 128, 0, stream>>>(p_w1, p_b1, p_w2, p_b2, per);
  k_temb<<<(B_ * T_ * D_ + 255) / 256, 256, 0, stream>>>(time_w, time_b, Temb);
  k_cemb<<<(B_ * C_ * D_ + 255) / 256, 256, 0, stream>>>(chan_w, chan_b, Cemb);
  {
    long n = (long)B_ * K_ * D_;
    k_u0<<<(unsigned)((n + 255) / 256), 256, 0, stream>>>(x, per, edge_w, edge_b, U);
  }

  auto run_side = [&](int l, int s) {
    int Nq = (s == 0) ? C_ : T_;
    const float* qsrc  = (s == 0) ? Cemb : Temb;
    const float* other = (s == 0) ? Temb : Cemb;
    float* qout = (s == 0) ? qcb : qtb;
    int ls = l * 2 + s;
    long nCat = (long)B_ * K_ * 256;
    k_concat2<<<(unsigned)((nCat + 255) / 256), 256, 0, stream>>>(U, other, big16, s);
    dim3 gK(B_ * K_ / 16, D_ / 64);
    k_gemm16<<<gK, 32, 0, stream>>>(big16, kwt(l, s), att_k_b + ls * D_, nullptr, ksb, B_ * K_, 256, D_, 0);
    k_gemm16<<<gK, 32, 0, stream>>>(big16, vwt(l, s), att_v_b + ls * D_, nullptr, vsb, B_ * K_, 256, D_, 0);
    int nq = B_ * Nq * D_;
    k_f32_to_f16<<<(nq + 255) / 256, 256, 0, stream>>>(qsrc, small16, nq);
    dim3 gQ(B_ * Nq / 16, D_ / 64);
    k_gemm16<<<gQ, 32, 0, stream>>>(small16, qwt(l, s), att_q_b + ls * D_, nullptr, qproj, B_ * Nq, D_, D_, 0);
    if (s == 0) k_attn_c<<<B_ * C_, 128, 0, stream>>>(qproj, ksb, vsb, attO);
    else        k_attn_t<<<B_ * T_, 128, 0, stream>>>(qproj, ksb, vsb, attO);
    k_ln<<<B_ * Nq, 128, 0, stream>>>(attO, ln0_g + ls * D_, ln0_b + ls * D_, attO);
    k_f32_to_f16<<<(nq + 255) / 256, 256, 0, stream>>>(attO, small16, nq);
    // X = O + relu(O @ ow + ob)  (fused relu + residual in GEMM epilogue)
    k_gemm16<<<gQ, 32, 0, stream>>>(small16, owt(l, s), att_o_b + ls * D_, attO, attO, B_ * Nq, D_, D_, 1);
    k_ln<<<B_ * Nq, 128, 0, stream>>>(attO, ln1_g + ls * D_, ln1_b + ls * D_, qout);
  };

  for (int l = 0; l < NL_; ++l) {
    run_side(l, 0);
    run_side(l, 1);
    long nC3 = (long)B_ * K_ * 384;
    k_concat3<<<(unsigned)((nC3 + 255) / 256), 256, 0, stream>>>(U, qtb, qcb, big16);
    dim3 gE(B_ * K_ / 16, D_ / 64);
    // U = relu(e_in @ W + b) + U   (fused, in place: each lane reads then writes its own element)
    k_gemm16<<<gE, 32, 0, stream>>>(big16, edget + (size_t)l * 49152, edge_nn_b + l * D_, U, U, B_ * K_, 384, D_, 1);
    hipMemcpyAsync(Cemb, qcb, sizeof(float) * (size_t)B_ * C_ * D_, hipMemcpyDeviceToDevice, stream);
    hipMemcpyAsync(Temb, qtb, sizeof(float) * (size_t)B_ * T_ * D_, hipMemcpyDeviceToDevice, stream);
  }

  k_out<<<(B_ * K_ + 127) / 128, 128, 0, stream>>>(U, Temb, Cemb, out_w, out_b, per, (float*)d_out);
}
